// TransformerBlock_16260746183244
// MI455X (gfx1250) — compile-verified, matrix operand order checked by reading, not run on hardware
//
#include <hip/hip_runtime.h>
#include <hip/hip_bf16.h>
#include <math.h>

#define D      1024
#define NHEAD  16
#define HDIM   64
#define HIDDEN 2752
#define NTOK   16384   // 4*16*256 tokens
#define SEQ    256
#define QKVW   (3*D)   // qkv row width

typedef __attribute__((ext_vector_type(16))) __bf16 bf16x16;
typedef __attribute__((ext_vector_type(8)))  float  f32x8;

// ---------- helpers ----------
__device__ __forceinline__ unsigned short f2bf(float f) {
    union { float f; unsigned u; } v; v.f = f;
    unsigned r = v.u + 0x7fffu + ((v.u >> 16) & 1u);   // round-to-nearest-even
    return (unsigned short)(r >> 16);
}

__device__ __forceinline__ float fast_rcp(float x) {
#if __has_builtin(__builtin_amdgcn_rcpf)
    return __builtin_amdgcn_rcpf(x);
#else
    return 1.0f / x;
#endif
}

__device__ __forceinline__ float fast_rsqrt(float x) {
#if __has_builtin(__builtin_amdgcn_rsqf)
    return __builtin_amdgcn_rsqf(x);
#else
    return rsqrtf(x);
#endif
}

__device__ __forceinline__ f32x8 wmma_bf16(bf16x16 a, bf16x16 b, f32x8 c) {
    return __builtin_amdgcn_wmma_f32_16x16x32_bf16(false, a, false, b, (short)0, c, false, false);
}

__device__ __forceinline__ f32x8 zero8() {
    f32x8 z;
#pragma unroll
    for (int i = 0; i < 8; ++i) z[i] = 0.0f;
    return z;
}

// ---------- async global->LDS copy (CDNA5), hedged with sync fallback ----------
#define ASYNC_G2L_OK __has_builtin(__builtin_amdgcn_global_load_async_to_lds_b128)

typedef int v4i __attribute__((ext_vector_type(4)));
typedef __attribute__((address_space(1))) v4i gas_v4i;   // global v4i32
typedef __attribute__((address_space(3))) v4i las_v4i;   // LDS v4i32

__device__ __forceinline__ void async_g2l_b128(void* lds, const void* gsrc) {
#if ASYNC_G2L_OK
    __builtin_amdgcn_global_load_async_to_lds_b128(
        (gas_v4i*)(unsigned long long)(uintptr_t)gsrc,
        (las_v4i*)(unsigned)(uintptr_t)lds, 0, 0);
#else
    *(uint4*)lds = *(const uint4*)gsrc;
#endif
}

__device__ __forceinline__ void wait_async() {
#if __has_builtin(__builtin_amdgcn_s_wait_asynccnt)
    __builtin_amdgcn_s_wait_asynccnt(0);
#else
    asm volatile("s_wait_asynccnt 0x0" ::: "memory");
#endif
}

// ---------- fp32 -> bf16 convert (plain) ----------
__global__ __launch_bounds__(256) void cvt_bf16_kernel(const float* __restrict__ src,
                                                       unsigned short* __restrict__ dst, int n) {
    int i = blockIdx.x * 256 + threadIdx.x;
    if (i < n) dst[i] = f2bf(src[i]);
}

// ---------- fp32 KxN -> bf16 NxK transpose-convert (weights, once per launch) ----------
__global__ __launch_bounds__(256) void cvt_tr_kernel(const float* __restrict__ src,
                                                     unsigned short* __restrict__ dst,
                                                     int K, int N) {
    __shared__ float t[32][33];
    const int tid = threadIdx.x;
    const int tx = tid & 31, ty = tid >> 5;          // 32 x 8 threads
    const int nb = blockIdx.x * 32, kb = blockIdx.y * 32;
#pragma unroll
    for (int i = 0; i < 4; ++i)
        t[ty + i * 8][tx] = src[(size_t)(kb + ty + i * 8) * N + nb + tx];
    __syncthreads();
#pragma unroll
    for (int i = 0; i < 4; ++i)
        dst[(size_t)(nb + ty + i * 8) * K + kb + tx] = f2bf(t[tx][ty + i * 8]);
}

// ---------- mod = silu(time_emb) @ mod_w + mod_b  (4 x 6144) ----------
__global__ __launch_bounds__(256) void mod_kernel(const float* __restrict__ te,
                                                  const float* __restrict__ mw,
                                                  const float* __restrict__ mb,
                                                  float* __restrict__ mod) {
    __shared__ float ste[D];
    const int b = blockIdx.y;
    const int tid = threadIdx.x;
#pragma unroll
    for (int i = 0; i < 4; ++i) {
        float v = te[b * D + tid + i * 256];
        ste[tid + i * 256] = v * fast_rcp(1.0f + __expf(-v));
    }
    __syncthreads();
    const int col = blockIdx.x * 256 + tid;
    float acc = mb[col];
    for (int k = 0; k < D; ++k) acc = fmaf(ste[k], mw[k * 6 * D + col], acc);
    mod[b * 6 * D + col] = acc;
}

// ---------- rmsnorm + (1+scale)*h + shift  ->  bf16 ----------
__global__ __launch_bounds__(256) void norm_mod_kernel(const float* __restrict__ x,
                                                       const float* __restrict__ nw,
                                                       const float* __restrict__ mod,
                                                       unsigned short* __restrict__ out,
                                                       int scale_off, int shift_off) {
    const int token = blockIdx.x;
    const int tid = threadIdx.x;
    const int lane = tid & 31, wid = tid >> 5;
    const int b = token >> 12;                       // token / 4096
    const float* xr = x + (size_t)token * D;
    float xv[4]; float ss = 0.0f;
#pragma unroll
    for (int i = 0; i < 4; ++i) { xv[i] = xr[tid + i * 256]; ss = fmaf(xv[i], xv[i], ss); }
#pragma unroll
    for (int off = 1; off < 32; off <<= 1) ss += __shfl_xor(ss, off, 32);
    __shared__ float red[8];
    if (lane == 0) red[wid] = ss;
    __syncthreads();
    float tot = 0.0f;
#pragma unroll
    for (int w = 0; w < 8; ++w) tot += red[w];
    const float rinv = fast_rsqrt(tot * (1.0f / D) + 1e-6f);
    const float* mrow = mod + (size_t)b * 6 * D;
#pragma unroll
    for (int i = 0; i < 4; ++i) {
        int col = tid + i * 256;
        float h = xv[i] * rinv * nw[col] * (1.0f + mrow[scale_off + col]) + mrow[shift_off + col];
        out[(size_t)token * D + col] = f2bf(h);
    }
}

// ---------- generic bf16 WMMA GEMM: C(MxN) = A(MxK) @ Bt(NxK)^T ----------
// B is PRE-TRANSPOSED to [N][K] so all staging is contiguous 128-bit copies.
// EPI 0: store f32   EPI 1: store bf16   EPI 2: f32 out = res + gate*acc
template <int EPI>
__global__ __launch_bounds__(256) void gemm_bf16_kernel(const unsigned short* __restrict__ A,
                                                        const unsigned short* __restrict__ Bt,
                                                        int M, int N, int K,
                                                        float* __restrict__ Cf,
                                                        unsigned short* __restrict__ Cbf,
                                                        const float* __restrict__ res,
                                                        const float* __restrict__ mod,
                                                        int gate_off) {
    __shared__ alignas(16) unsigned short As[2][128][32];
    __shared__ alignas(16) unsigned short Bs[2][128][32];
    const int tid = threadIdx.x;
    const int lane = tid & 31, wid = tid >> 5;
    const int l16 = lane & 15, half = lane >> 4;
    const int wm = wid & 3, wn = wid >> 2;           // 4x2 wave grid, 32x64 per wave
    const int m0 = blockIdx.y * 128, n0 = blockIdx.x * 128;
    const int srow = tid >> 1, sseg = (tid & 1) * 16;

    auto stage = [&](int buf, int k0) {
        const unsigned short* sa = A  + (size_t)(m0 + srow) * K + k0 + sseg;
        const unsigned short* sb = Bt + (size_t)(n0 + srow) * K + k0 + sseg;
        async_g2l_b128(&As[buf][srow][sseg],     sa);
        async_g2l_b128(&As[buf][srow][sseg + 8], sa + 8);
        async_g2l_b128(&Bs[buf][srow][sseg],     sb);
        async_g2l_b128(&Bs[buf][srow][sseg + 8], sb + 8);
    };

    f32x8 acc[2][4];
#pragma unroll
    for (int mt = 0; mt < 2; ++mt)
#pragma unroll
        for (int nt = 0; nt < 4; ++nt) acc[mt][nt] = zero8();

    stage(0, 0);
    wait_async();
    __syncthreads();

    const int nk = K >> 5;
    for (int ki = 0; ki < nk; ++ki) {
        const int cur = ki & 1;
        if (ki + 1 < nk) stage(cur ^ 1, (ki + 1) << 5);   // copy-in overlaps WMMA

        bf16x16 af[2], bfr[4];
#pragma unroll
        for (int mt = 0; mt < 2; ++mt)
            af[mt] = *(const bf16x16*)&As[cur][wm * 32 + mt * 16 + l16][half * 16];
#pragma unroll
        for (int nt = 0; nt < 4; ++nt)
            bfr[nt] = *(const bf16x16*)&Bs[cur][wn * 64 + nt * 16 + l16][half * 16];
#pragma unroll
        for (int mt = 0; mt < 2; ++mt)
#pragma unroll
            for (int nt = 0; nt < 4; ++nt)
                acc[mt][nt] = wmma_bf16(af[mt], bfr[nt], acc[mt][nt]);

        wait_async();
        __syncthreads();
    }

#pragma unroll
    for (int mt = 0; mt < 2; ++mt)
#pragma unroll
        for (int nt = 0; nt < 4; ++nt)
#pragma unroll
            for (int j = 0; j < 8; ++j) {
                int grow = m0 + wm * 32 + mt * 16 + j + half * 8;
                int gcol = n0 + wn * 64 + nt * 16 + l16;
                float v = acc[mt][nt][j];
                size_t idx = (size_t)grow * N + gcol;
                if (EPI == 0) Cf[idx] = v;
                else if (EPI == 1) Cbf[idx] = f2bf(v);
                else {
                    float g = mod[(size_t)(grow >> 12) * 6 * D + gate_off + gcol];
                    Cf[idx] = res[idx] + g * v;
                }
            }
}

// ---------- fused MLP gate GEMM: g = bf16( silu(A@W1) * (A@W2) ), W pre-transposed ----------
__global__ __launch_bounds__(256) void mlp_gate_kernel(const unsigned short* __restrict__ A,
                                                       const unsigned short* __restrict__ W1t,
                                                       const unsigned short* __restrict__ W2t,
                                                       int M, int N, int K,
                                                       unsigned short* __restrict__ G) {
    __shared__ alignas(16) unsigned short As[2][128][32];
    __shared__ alignas(16) unsigned short B1[2][64][32];
    __shared__ alignas(16) unsigned short B2[2][64][32];
    const int tid = threadIdx.x;
    const int lane = tid & 31, wid = tid >> 5;
    const int l16 = lane & 15, half = lane >> 4;
    const int wm = wid & 3, wn = wid >> 2;           // 4x2 waves, 32x32 per wave
    const int m0 = blockIdx.y * 128, n0 = blockIdx.x * 64;
    const int arow = tid >> 1, aseg = (tid & 1) * 16;
    const int brow = tid >> 2, bseg = (tid & 3) * 8;

    auto stage = [&](int buf, int k0) {
        const unsigned short* sa = A   + (size_t)(m0 + arow) * K + k0 + aseg;
        const unsigned short* s1 = W1t + (size_t)(n0 + brow) * K + k0 + bseg;
        const unsigned short* s2 = W2t + (size_t)(n0 + brow) * K + k0 + bseg;
        async_g2l_b128(&As[buf][arow][aseg],     sa);
        async_g2l_b128(&As[buf][arow][aseg + 8], sa + 8);
        async_g2l_b128(&B1[buf][brow][bseg], s1);
        async_g2l_b128(&B2[buf][brow][bseg], s2);
    };

    f32x8 a1[2][2], a2[2][2];
#pragma unroll
    for (int mt = 0; mt < 2; ++mt)
#pragma unroll
        for (int nt = 0; nt < 2; ++nt) { a1[mt][nt] = zero8(); a2[mt][nt] = zero8(); }

    stage(0, 0);
    wait_async();
    __syncthreads();

    const int nk = K >> 5;
    for (int ki = 0; ki < nk; ++ki) {
        const int cur = ki & 1;
        if (ki + 1 < nk) stage(cur ^ 1, (ki + 1) << 5);

        bf16x16 af[2], b1f[2], b2f[2];
#pragma unroll
        for (int mt = 0; mt < 2; ++mt)
            af[mt] = *(const bf16x16*)&As[cur][wm * 32 + mt * 16 + l16][half * 16];
#pragma unroll
        for (int nt = 0; nt < 2; ++nt) {
            b1f[nt] = *(const bf16x16*)&B1[cur][wn * 32 + nt * 16 + l16][half * 16];
            b2f[nt] = *(const bf16x16*)&B2[cur][wn * 32 + nt * 16 + l16][half * 16];
        }
#pragma unroll
        for (int mt = 0; mt < 2; ++mt)
#pragma unroll
            for (int nt = 0; nt < 2; ++nt) {
                a1[mt][nt] = wmma_bf16(af[mt], b1f[nt], a1[mt][nt]);
                a2[mt][nt] = wmma_bf16(af[mt], b2f[nt], a2[mt][nt]);
            }

        wait_async();
        __syncthreads();
    }

#pragma unroll
    for (int mt = 0; mt < 2; ++mt)
#pragma unroll
        for (int nt = 0; nt < 2; ++nt)
#pragma unroll
            for (int j = 0; j < 8; ++j) {
                int grow = m0 + wm * 32 + mt * 16 + j + half * 8;
                int gcol = n0 + wn * 32 + nt * 16 + l16;
                float u = a1[mt][nt][j];
                float sig = u * fast_rcp(1.0f + __expf(-u));
                G[(size_t)grow * N + gcol] = f2bf(sig * a2[mt][nt][j]);
            }
}

// ---------- flash attention, one (head, b*t) per block, 8 waves x 32 rows ----------
__global__ __launch_bounds__(256) void attn_kernel(const unsigned short* __restrict__ qkv,
                                                   unsigned short* __restrict__ out) {
    __shared__ alignas(16) unsigned short Kt[32][64];     // [keyLocal][d]
    __shared__ alignas(16) unsigned short Vt[64][32];     // [d][keyLocal] (transposed)
    __shared__ alignas(16) unsigned short Pb[8][32][32];  // per-wave P bounce buffer
    const int head = blockIdx.x;
    const int bt   = blockIdx.y;
    const int tid = threadIdx.x;
    const int lane = tid & 31, wid = tid >> 5;
    const int l16 = lane & 15, half = lane >> 4;
    const int tokBase = bt * SEQ;
    const float scale = 0.125f;                           // 1/sqrt(64)

    // Q fragments straight from global qkv (bf16), 2 m-tiles x 2 K-chunks
    bf16x16 qf[2][2];
#pragma unroll
    for (int mt = 0; mt < 2; ++mt) {
        int qrow = tokBase + wid * 32 + mt * 16 + l16;
        const unsigned short* qp = qkv + (size_t)qrow * QKVW + head * HDIM + half * 16;
        qf[mt][0] = *(const bf16x16*)(qp);
        qf[mt][1] = *(const bf16x16*)(qp + 32);
    }

    float m_run[2][8], l_run[2][8];
    f32x8 o[2][4];
#pragma unroll
    for (int mt = 0; mt < 2; ++mt) {
#pragma unroll
        for (int j = 0; j < 8; ++j) { m_run[mt][j] = -1e30f; l_run[mt][j] = 0.0f; }
#pragma unroll
        for (int dt = 0; dt < 4; ++dt) o[mt][dt] = zero8();
    }

    for (int kp = 0; kp < SEQ / 32; ++kp) {
        __syncthreads();
        {   // stage 32 keys: K rows async, V rows transposed via LDS scatter
            int key = tid >> 3;                 // 0..31
            int d0 = (tid & 7) * 8;             // 0..56
            const unsigned short* kpp = qkv + (size_t)(tokBase + kp * 32 + key) * QKVW + D + head * HDIM + d0;
            const unsigned short* vpp = qkv + (size_t)(tokBase + kp * 32 + key) * QKVW + 2 * D + head * HDIM + d0;
            async_g2l_b128(&Kt[key][d0], kpp);
            uint4 vv = *(const uint4*)vpp;
            const unsigned short* vs = (const unsigned short*)&vv;
#pragma unroll
            for (int j = 0; j < 8; ++j) Vt[d0 + j][key] = vs[j];
        }
        wait_async();
        __syncthreads();

        // S = Q @ K^T  (2 m-tiles x 2 key-tiles), K-dim = 64 via two WMMA
        f32x8 s[2][2];
#pragma unroll
        for (int mt = 0; mt < 2; ++mt)
#pragma unroll
            for (int nt = 0; nt < 2; ++nt) {
                f32x8 a = zero8();
#pragma unroll
                for (int kc = 0; kc < 2; ++kc) {
                    bf16x16 kf = *(const bf16x16*)&Kt[nt * 16 + l16][kc * 32 + half * 16];
                    a = wmma_bf16(qf[mt][kc], kf, a);
                }
                s[mt][nt] = a;
            }

        // online softmax; row r of a 16x16 C tile lives in one 16-lane half
#pragma unroll
        for (int mt = 0; mt < 2; ++mt) {
#pragma unroll
            for (int j = 0; j < 8; ++j) {
                float s0 = s[mt][0][j] * scale;
                float s1 = s[mt][1][j] * scale;
                float mx = fmaxf(s0, s1);
#pragma unroll
                for (int off = 1; off < 16; off <<= 1) mx = fmaxf(mx, __shfl_xor(mx, off, 32));
                float mnew = fmaxf(m_run[mt][j], mx);
                float alpha = __expf(m_run[mt][j] - mnew);
                float p0 = __expf(s0 - mnew);
                float p1 = __expf(s1 - mnew);
                float ps = p0 + p1;
#pragma unroll
                for (int off = 1; off < 16; off <<= 1) ps += __shfl_xor(ps, off, 32);
                l_run[mt][j] = l_run[mt][j] * alpha + ps;
                m_run[mt][j] = mnew;
#pragma unroll
                for (int dt = 0; dt < 4; ++dt) o[mt][dt][j] *= alpha;
                int r = mt * 16 + j + half * 8;
                Pb[wid][r][l16]      = f2bf(p0);
                Pb[wid][r][16 + l16] = f2bf(p1);
            }
        }
        asm volatile("s_wait_dscnt 0" ::: "memory");   // wave-private LDS RAW

        // O += P @ V   (K=32 keys per step, 4 d-tiles)
#pragma unroll
        for (int mt = 0; mt < 2; ++mt) {
            bf16x16 pf = *(const bf16x16*)&Pb[wid][mt * 16 + l16][half * 16];
#pragma unroll
            for (int dt = 0; dt < 4; ++dt) {
                bf16x16 vf = *(const bf16x16*)&Vt[dt * 16 + l16][half * 16];
                o[mt][dt] = wmma_bf16(pf, vf, o[mt][dt]);
            }
        }
    }

    // write normalized output (row-major tokens x 1024, bf16)
#pragma unroll
    for (int mt = 0; mt < 2; ++mt) {
        float rl[8];
#pragma unroll
        for (int j = 0; j < 8; ++j) rl[j] = fast_rcp(l_run[mt][j]);
#pragma unroll
        for (int dt = 0; dt < 4; ++dt)
#pragma unroll
            for (int j = 0; j < 8; ++j) {
                int row = tokBase + wid * 32 + mt * 16 + j + half * 8;
                int col = head * HDIM + dt * 16 + l16;
                out[(size_t)row * D + col] = f2bf(o[mt][dt][j] * rl[j]);
            }
    }
}

// ---------- host ----------
extern "C" void kernel_launch(void* const* d_in, const int* in_sizes, int n_in,
                              void* d_out, int out_size, void* d_ws, size_t ws_size,
                              hipStream_t stream) {
    const float* x     = (const float*)d_in[0];
    const float* te    = (const float*)d_in[1];
    const float* n1w   = (const float*)d_in[2];
    const float* n2w   = (const float*)d_in[3];
    const float* qkvw  = (const float*)d_in[4];
    const float* outw  = (const float*)d_in[5];
    const float* w1    = (const float*)d_in[6];
    const float* w2    = (const float*)d_in[7];
    const float* w3    = (const float*)d_in[8];
    const float* modw  = (const float*)d_in[9];
    const float* modb  = (const float*)d_in[10];
    float* out = (float*)d_out;

    char* ws = (char*)d_ws;
    size_t off = 0;
    auto alloc = [&](size_t bytes) { void* p = ws + off; off += (bytes + 255) & ~(size_t)255; return p; };

    float*          modv   = (float*)alloc(4 * 6 * D * sizeof(float));
    unsigned short* WqT    = (unsigned short*)alloc((size_t)3 * D * D * 2);     // [3D][D]
    unsigned short* WoT    = (unsigned short*)alloc((size_t)D * D * 2);         // [D][D]
    unsigned short* W1T    = (unsigned short*)alloc((size_t)HIDDEN * D * 2);    // [HIDDEN][D]
    unsigned short* W2T    = (unsigned short*)alloc((size_t)HIDDEN * D * 2);
    unsigned short* W3T    = (unsigned short*)alloc((size_t)D * HIDDEN * 2);    // [D][HIDDEN]
    unsigned short* h1     = (unsigned short*)alloc((size_t)NTOK * D * 2);
    unsigned short* qkvbuf = (unsigned short*)alloc((size_t)NTOK * 3 * D * 2);
    unsigned short* attn   = (unsigned short*)alloc((size_t)NTOK * D * 2);
    float*          xmid   = (float*)alloc((size_t)NTOK * D * sizeof(float));
    unsigned short* h2     = (unsigned short*)alloc((size_t)NTOK * D * 2);
    unsigned short* g      = (unsigned short*)alloc((size_t)NTOK * HIDDEN * 2);

    // 1. weight conversion to bf16 with transpose: src [K][N] -> dst [N][K]
    auto cvt_tr = [&](const float* s, unsigned short* d, int K, int N) {
        cvt_tr_kernel<<<dim3(N / 32, K / 32), 256, 0, stream>>>(s, d, K, N);
    };
    cvt_tr(qkvw, WqT, D, 3 * D);
    cvt_tr(outw, WoT, D, D);
    cvt_tr(w1, W1T, D, HIDDEN);
    cvt_tr(w2, W2T, D, HIDDEN);
    cvt_tr(w3, W3T, HIDDEN, D);

    // 2. modulation vector
    mod_kernel<<<dim3(6 * D / 256, 4), 256, 0, stream>>>(te, modw, modb, modv);

    // 3. rmsnorm1 + modulate (shift1=0, scale1=D)
    norm_mod_kernel<<<NTOK, 256, 0, stream>>>(x, n1w, modv, h1, /*scale*/ D, /*shift*/ 0);

    // 4. qkv GEMM -> bf16
    gemm_bf16_kernel<1><<<dim3(3 * D / 128, NTOK / 128), 256, 0, stream>>>(
        h1, WqT, NTOK, 3 * D, D, nullptr, qkvbuf, nullptr, nullptr, 0);

    // 5. flash attention
    attn_kernel<<<dim3(NHEAD, 64), 256, 0, stream>>>(qkvbuf, attn);

    // 6. out-proj GEMM + residual with gate1 (offset 2D) -> xmid
    gemm_bf16_kernel<2><<<dim3(D / 128, NTOK / 128), 256, 0, stream>>>(
        attn, WoT, NTOK, D, D, xmid, nullptr, x, modv, 2 * D);

    // 7. rmsnorm2 + modulate (shift2=3D, scale2=4D)
    norm_mod_kernel<<<NTOK, 256, 0, stream>>>(xmid, n2w, modv, h2, 4 * D, 3 * D);

    // 8. fused silu(h@W1)*(h@W2) -> g (bf16)
    mlp_gate_kernel<<<dim3(HIDDEN / 64, NTOK / 128), 256, 0, stream>>>(
        h2, W1T, W2T, NTOK, HIDDEN, D, g);

    // 9. g @ W3 + residual with gate2 (offset 5D) -> final output
    gemm_bf16_kernel<2><<<dim3(D / 128, NTOK / 128), 256, 0, stream>>>(
        g, W3T, NTOK, D, HIDDEN, out, nullptr, xmid, modv, 5 * D);

    (void)in_sizes; (void)n_in; (void)out_size; (void)ws_size;
}
